// LiquidRON_84516366450723
// MI455X (gfx1250) — matrix operand order repaired and checked by reading, not compiled, not run on hardware
//
#include <hip/hip_runtime.h>

// ---------------------------------------------------------------------------
// Izhikevich reservoir, persistent 8-workgroup kernel for MI455X (gfx1250).
//  - S (1000x1000 f32) converted once to f16 and kept LDS-resident:
//      8 WGs x 128 rows x 1024 padded cols x 2B = 256 KB/WGP (of 320 KB).
//  - Per step: each wave computes y[16 rows] = S_rows @ fired via
//      32 x v_wmma_f32_16x16x32_f16 with the mask chunk replicated across
//      all 16 B columns (every D column == y partial).
//  - Cross-WG mask exchange: 2 KB f16 double buffer in global memory,
//      monotonic atomic barrier (+ s_cluster_barrier, NOP if unclustered).
// ---------------------------------------------------------------------------

typedef __attribute__((ext_vector_type(16))) _Float16 v16h;
typedef __attribute__((ext_vector_type(8)))  _Float16 v8h;
typedef __attribute__((ext_vector_type(8)))  float    v8f;

#define NN      1000      // real neurons
#define NP      1024      // padded
#define TT      16384
#define NWG     8
#define ROWS_WG 128       // NP / NWG
#define THREADS 256       // 8 wave32
#define KTILES  32        // NP / 32

#define S_BYTES  (ROWS_WG * NP * 2)      // 262144
#define FM_BYTES (NP * 2)                // 2048
#define Y_BYTES  (8 * 16 * 4)            // 512
#define SHMEM    (S_BYTES + FM_BYTES + Y_BYTES)

__global__ void __launch_bounds__(THREADS, 1)
izh_reservoir_kernel(const float* __restrict__ data, const float* __restrict__ U,
                     const float* __restrict__ S,    const float* __restrict__ a,
                     const float* __restrict__ b,    const float* __restrict__ c,
                     const float* __restrict__ d,    const float* __restrict__ v0,
                     const float* __restrict__ u0,   float* __restrict__ out,
                     unsigned* __restrict__ barCnt,  _Float16* __restrict__ fmbuf)
{
    extern __shared__ char smem[];
    _Float16* Ssh  = (_Float16*)smem;                       // [128][1024] f16
    _Float16* fmsh = (_Float16*)(smem + S_BYTES);           // [1024] f16
    float*    ysh  = (float*)(smem + S_BYTES + FM_BYTES);   // [8][16] f32

    const int tid   = threadIdx.x;
    const int wg    = blockIdx.x;
    const int wave  = tid >> 5;
    const int lane  = tid & 31;
    const int rowIW = lane & 15;            // A/D row within wave tile
    const int ahalf = (lane >> 4) * 16;     // A K-half: {0..15} vs {16..31}
    const int bhalf = (lane >> 4) * 16;     // B K-half: {0..15} vs {16..31}

    // ---- one-time: stage this WG's 128 S rows into LDS as f16 (zero-padded)
    for (int idx = tid; idx < ROWS_WG * NP; idx += THREADS) {
        int r = idx >> 10, col = idx & (NP - 1);
        int gr = wg * ROWS_WG + r;
        float sv = (gr < NN && col < NN) ? S[gr * NN + col] : 0.0f;
        Ssh[idx] = (_Float16)sv;
    }

    // ---- per-neuron fp32 state in registers: lanes 0..15 of each wave own 16 neurons
    const int  n   = wg * ROWS_WG + wave * 16 + rowIW;   // 0..1023
    const bool own = (lane < 16) && (n < NN);
    const int  ln  = own ? n : 0;
    const float pa = a[ln], pb = b[ln], pc = c[ln], pd = d[ln], pU = U[ln];
    float v = v0[ln], u = u0[ln];

    __syncthreads();

    unsigned gen = 0;
    const size_t TN = (size_t)TT * NN;

    for (int t = 0; t < TT; ++t) {
        // -- phase 1: fired mask from current state; publish to global double buffer
        float fm = 0.0f;
        if (lane < 16) {
            bool fired = (v >= 30.0f);
            fm = fired ? 1.0f : 0.0f;
            v  = fired ? pc : v;           // reset (idempotent double-apply in ref)
            u  = u + 2.0f * pd * fm;       // duplicated update in ref -> 2*d
            if (own) {
                fmbuf[(t & 1) * NP + n] = (_Float16)fm;
                out[TN + 2 * NN + (size_t)t * NN + n] = fm;   // firings[t]
            }
        }
        __threadfence();                   // release fm stores to device scope
        __syncthreads();                   // all lanes' stores precede arrival

        // -- phase 2: 8-WG barrier. Cluster-barrier HW path (NOP if unclustered)
        //    + monotonic atomic counter as the architected-independent sync.
        if (wave == 0) __builtin_amdgcn_s_cluster_barrier();
        if (tid == 0) {
            __hip_atomic_fetch_add(barCnt, 1u, __ATOMIC_RELEASE,
                                   __HIP_MEMORY_SCOPE_AGENT);
            const unsigned target = (gen + 1u) * (unsigned)NWG;
            while (__hip_atomic_load(barCnt, __ATOMIC_ACQUIRE,
                                     __HIP_MEMORY_SCOPE_AGENT) < target) {
                __builtin_amdgcn_s_sleep(1);
            }
        }
        gen++;
        __syncthreads();
        __threadfence();                   // acquire: discard stale cached mask

        // -- phase 3: pull full 1024-entry mask into LDS (2 KB, 8 B per thread)
        ((unsigned long long*)fmsh)[tid] =
            ((const unsigned long long*)(fmbuf + (t & 1) * NP))[tid];
        __syncthreads();

        // -- phase 4: y = S_rows @ fm via WMMA, two independent accumulator chains
        v8f acc0 = {}, acc1 = {};
        const _Float16* arow = Ssh + (wave * 16 + rowIW) * NP;
        #pragma unroll 4
        for (int kt = 0; kt < KTILES; kt += 2) {
            {
                const _Float16* ap = arow + kt * 32;
                v8h alo = *(const v8h*)(ap + ahalf);
                v8h ahi = *(const v8h*)(ap + 8 + ahalf);
                union { v16h w; v8h h[2]; } au; au.h[0] = alo; au.h[1] = ahi;
                v16h bv = *(const v16h*)(fmsh + kt * 32 + bhalf);
                acc0 = __builtin_amdgcn_wmma_f32_16x16x32_f16(
                           false, au.w, false, bv, (short)0, acc0, false, false);
            }
            {
                const _Float16* ap = arow + (kt + 1) * 32;
                v8h alo = *(const v8h*)(ap + ahalf);
                v8h ahi = *(const v8h*)(ap + 8 + ahalf);
                union { v16h w; v8h h[2]; } au; au.h[0] = alo; au.h[1] = ahi;
                v16h bv = *(const v16h*)(fmsh + (kt + 1) * 32 + bhalf);
                acc1 = __builtin_amdgcn_wmma_f32_16x16x32_f16(
                           false, au.w, false, bv, (short)0, acc1, false, false);
            }
        }
        v8f acc = acc0 + acc1;

        // -- phase 5: D columns are all equal; lane 0 holds y[0..7], lane 16 y[8..15]
        if ((lane & 15) == 0) {
            float* yp = ysh + wave * 16 + (lane >> 4) * 8;
            #pragma unroll
            for (int r = 0; r < 8; ++r) yp[r] = acc[r];
        }
        // intra-wave LDS RAW: compiler inserts s_wait_dscnt

        // -- phase 6: neuron dynamics + states output
        if (lane < 16) {
            float y  = ysh[wave * 16 + rowIW];
            float xI = data[t] * pU + y;
            v = v + 0.5f * (0.04f * v * v + 5.0f * v + 140.0f - u + xI);
            u = u + pa * (pb * v - u);
            if (own) out[(size_t)t * NN + n] = (v >= 30.0f) ? 1.0f : 0.0f;
        }
    }

    if (own) {
        out[TN + n]      = v;
        out[TN + NN + n] = u;
    }
}

extern "C" void kernel_launch(void* const* d_in, const int* in_sizes, int n_in,
                              void* d_out, int out_size, void* d_ws, size_t ws_size,
                              hipStream_t stream) {
    const float* data = (const float*)d_in[0];
    const float* U    = (const float*)d_in[1];
    const float* S    = (const float*)d_in[2];
    const float* a    = (const float*)d_in[3];
    const float* b    = (const float*)d_in[4];
    const float* c    = (const float*)d_in[5];
    const float* d    = (const float*)d_in[6];
    const float* v0   = (const float*)d_in[7];
    const float* u0   = (const float*)d_in[8];
    float* out = (float*)d_out;

    // workspace: [0,64) barrier counter (monotonic), [64, 64+4096) fm double buffer
    unsigned*  barCnt = (unsigned*)d_ws;
    _Float16*  fmbuf  = (_Float16*)((char*)d_ws + 64);
    hipMemsetAsync(d_ws, 0, 64 + 2 * NP * sizeof(_Float16), stream);

    izh_reservoir_kernel<<<NWG, THREADS, SHMEM, stream>>>(
        data, U, S, a, b, c, d, v0, u0, out, barCnt, fmbuf);
}
